// BitLinear_23106924052906
// MI455X (gfx1250) — matrix operand order repaired and checked by reading, not compiled, not run on hardware
//
#include <hip/hip_runtime.h>
#include <cstdint>
#include <cstddef>

// ---------------- vector types ----------------
typedef __attribute__((ext_vector_type(16))) __bf16 v16bf;
typedef __attribute__((ext_vector_type(8)))  __bf16 v8bf;
typedef __attribute__((ext_vector_type(4)))  __bf16 v4bf;
typedef __attribute__((ext_vector_type(8)))  float  v8f;
typedef int v4i __attribute__((vector_size(16)));   // int4: matches async builtin signature

typedef __attribute__((address_space(1))) v4i gv4i;   // global int4
typedef __attribute__((address_space(3))) v4i lv4i;   // LDS int4

#define ALPHA 0.7f

// Problem shape fixed by reference setup_inputs()
static constexpr int MDIM = 4 * 2048;   // B*S
static constexpr int KDIM = 2048;       // IN
static constexpr int NDIM = 8192;       // OUT

// GEMM tiling: block = 128(M) x 256(N) x 32(K); 8 waves as 2x4, each wave 64x64
static constexpr int BM = 128, BN = 256, BK = 32;
static constexpr int LDT = BK + 8;          // padded LDS row: 40 bf16 = 80B = 20 banks -> conflict free
static constexpr int KTILES = KDIM / BK;    // 64

#if __has_builtin(__builtin_amdgcn_global_load_async_to_lds_b128)
#define HAVE_ASYNC_LDS 1
#else
#define HAVE_ASYNC_LDS 0
#endif

__device__ __forceinline__ void g2l_16B(void* lds_dst, const void* gsrc) {
#if HAVE_ASYNC_LDS
  __builtin_amdgcn_global_load_async_to_lds_b128(
      (gv4i*)(void*)gsrc,
      (lv4i*)lds_dst,
      0, 0);
#else
  *(v8bf*)lds_dst = *(const v8bf*)gsrc;   // fallback: global load + ds_store
#endif
}

__device__ __forceinline__ void async_wait0() {
#if HAVE_ASYNC_LDS
#if __has_builtin(__builtin_amdgcn_s_wait_asynccnt)
  __builtin_amdgcn_s_wait_asynccnt(0);
#else
  asm volatile("s_wait_asynccnt 0" ::: "memory");
#endif
#endif
}

// ---------------- stage 1: |W| partial sums (deterministic fixed-order) ----------------
__global__ void absmean_partial_kernel(const float* __restrict__ w,
                                       float* __restrict__ partials, int n4) {
  __shared__ float sm[256];
  const int t = threadIdx.x;
  const int stride = gridDim.x * 256;
  float s = 0.f;
  for (int i = blockIdx.x * 256 + t; i < n4; i += stride) {
    const float4 v = ((const float4*)w)[i];
    s += fabsf(v.x) + fabsf(v.y) + fabsf(v.z) + fabsf(v.w);
  }
  sm[t] = s;
  __syncthreads();
  for (int off = 128; off > 0; off >>= 1) {
    if (t < off) sm[t] += sm[t + off];
    __syncthreads();
  }
  if (t == 0) partials[blockIdx.x] = sm[0];
}

// ---------------- stage 2: final mean ----------------
__global__ void absmean_final_kernel(const float* __restrict__ partials, int nparts,
                                     float* __restrict__ scale_out, float inv_n) {
  __shared__ float sm[256];
  const int t = threadIdx.x;
  float s = 0.f;
  for (int i = t; i < nparts; i += 256) s += partials[i];
  sm[t] = s;
  __syncthreads();
  for (int off = 128; off > 0; off >>= 1) {
    if (t < off) sm[t] += sm[t + off];
    __syncthreads();
  }
  if (t == 0) *scale_out = sm[0] * inv_n;
}

// ---------------- ternary-quantize W -> bf16 {-1,0,+1} ----------------
__global__ void quantize_w_kernel(const float* __restrict__ w,
                                  const float* __restrict__ scale_p,
                                  __bf16* __restrict__ wq, int n4) {
  const int i = blockIdx.x * 256 + threadIdx.x;
  if (i >= n4) return;
  const float thr = ALPHA * (*scale_p);
  const float4 v = ((const float4*)w)[i];
  v4bf q;
  q.x = (__bf16)((fabsf(v.x) >= thr) ? (v.x > 0.f ? 1.f : -1.f) : 0.f);
  q.y = (__bf16)((fabsf(v.y) >= thr) ? (v.y > 0.f ? 1.f : -1.f) : 0.f);
  q.z = (__bf16)((fabsf(v.z) >= thr) ? (v.z > 0.f ? 1.f : -1.f) : 0.f);
  q.w = (__bf16)((fabsf(v.w) >= thr) ? (v.w > 0.f ? 1.f : -1.f) : 0.f);
  ((v4bf*)wq)[i] = q;
}

// ---------------- convert x fp32 -> bf16 ----------------
__global__ void convert_x_kernel(const float* __restrict__ x,
                                 __bf16* __restrict__ xb, int n4) {
  const int i = blockIdx.x * 256 + threadIdx.x;
  if (i >= n4) return;
  const float4 v = ((const float4*)x)[i];
  v4bf o;
  o.x = (__bf16)v.x; o.y = (__bf16)v.y; o.z = (__bf16)v.z; o.w = (__bf16)v.w;
  ((v4bf*)xb)[i] = o;
}

// ---------------- bf16 WMMA GEMM: out = (Xb @ Wq^T) * scale ----------------
__global__ void __launch_bounds__(256)
bitlinear_gemm_kernel(const __bf16* __restrict__ Xb,
                      const __bf16* __restrict__ Wq,
                      const float* __restrict__ scale_p,
                      float* __restrict__ Out) {
  __shared__ __bf16 As[2][BM * LDT];   // 2 x 128 x 40 bf16 = 20 KB
  __shared__ __bf16 Bs[2][BN * LDT];   // 2 x 256 x 40 bf16 = 40 KB

  const int tid  = threadIdx.x;
  const int lane = tid & 31;
  const int wave = tid >> 5;     // 0..7
  const int wm = wave >> 2;      // 0..1 -> 64-row slab
  const int wn = wave & 3;       // 0..3 -> 64-col slab

  const int m0 = blockIdx.y * BM;
  const int n0 = blockIdx.x * BN;

  // global->LDS tile fill: each thread moves 16B per chunk; 256 thr cover 64 rows x 32 cols
  const int lrow = tid >> 2;           // 0..63
  const int lcol = (tid & 3) * 8;      // 0,8,16,24 (bf16 elems)

  const float scale = *scale_p;

  v8f acc[4][4] = {};

  auto load_tiles = [&](int kt, int buf) {
    const size_t kbase = (size_t)kt * BK + lcol;
    const int loff = lrow * LDT + lcol;
    // A: 128 rows in 2 chunks
    g2l_16B(&As[buf][loff],             Xb + (size_t)(m0 + lrow) * KDIM + kbase);
    g2l_16B(&As[buf][loff + 64 * LDT],  Xb + (size_t)(m0 + lrow + 64) * KDIM + kbase);
    // B: 256 rows in 4 chunks
#pragma unroll
    for (int c = 0; c < 4; ++c)
      g2l_16B(&Bs[buf][loff + c * 64 * LDT],
              Wq + (size_t)(n0 + lrow + c * 64) * KDIM + kbase);
  };

  const int l15 = lane & 15;
  const int hi  = lane >> 4;   // half-wave select
  const int ka  = hi * 16;     // A-frag: lanes 0-15 K=0-15; lanes 16-31 K=16-31 (contiguous)
  const int kb  = hi * 16;     // B-frag: lanes 0-15 K=0-15; lanes 16-31 K=16-31 (contiguous)

  load_tiles(0, 0);
  async_wait0();
  __syncthreads();

  for (int kt = 0; kt < KTILES; ++kt) {
    const int buf = kt & 1;
    if (kt + 1 < KTILES) load_tiles(kt + 1, buf ^ 1);   // prefetch next tile (async)

    v16bf afrag[4], bfrag[4];
#pragma unroll
    for (int mt = 0; mt < 4; ++mt) {
      const __bf16* p = &As[buf][(wm * 64 + mt * 16 + l15) * LDT + ka];
      v8bf lo  = *(const v8bf*)p;
      v8bf hi8 = *(const v8bf*)(p + 8);
      afrag[mt] = __builtin_shufflevector(lo, hi8,
          0,1,2,3,4,5,6,7,8,9,10,11,12,13,14,15);
    }
#pragma unroll
    for (int nt = 0; nt < 4; ++nt) {
      const __bf16* p = &Bs[buf][(wn * 64 + nt * 16 + l15) * LDT + kb];
      v8bf lo  = *(const v8bf*)p;
      v8bf hi8 = *(const v8bf*)(p + 8);
      bfrag[nt] = __builtin_shufflevector(lo, hi8,
          0,1,2,3,4,5,6,7,8,9,10,11,12,13,14,15);
    }

#pragma unroll
    for (int mt = 0; mt < 4; ++mt)
#pragma unroll
      for (int nt = 0; nt < 4; ++nt)
        acc[mt][nt] = __builtin_amdgcn_wmma_f32_16x16x32_bf16(
            false, afrag[mt], false, bfrag[nt],
            (short)0, acc[mt][nt], false, false);

    async_wait0();     // prefetch for next iter landed
    __syncthreads();   // all waves done reading current buf
  }

  // D layout: VGPR r -> M = r + 8*(lane>=16), N = lane&15
#pragma unroll
  for (int mt = 0; mt < 4; ++mt) {
#pragma unroll
    for (int nt = 0; nt < 4; ++nt) {
      float* o = Out + (size_t)(m0 + wm * 64 + mt * 16 + hi * 8) * NDIM
                     + (n0 + wn * 64 + nt * 16 + l15);
#pragma unroll
      for (int r = 0; r < 8; ++r)
        o[(size_t)r * NDIM] = acc[mt][nt][r] * scale;
    }
  }
}

// ---------------- launcher ----------------
extern "C" void kernel_launch(void* const* d_in, const int* in_sizes, int n_in,
                              void* d_out, int out_size, void* d_ws, size_t ws_size,
                              hipStream_t stream) {
  const float* x = (const float*)d_in[0];   // (4,2048,2048) fp32
  const float* w = (const float*)d_in[1];   // (8192,2048) fp32
  float* out = (float*)d_out;               // (4,2048,8192) fp32

  char* ws = (char*)d_ws;
  __bf16* Xb = (__bf16*)ws;                                              // 32 MB
  __bf16* Wq = (__bf16*)(ws + (size_t)MDIM * KDIM * sizeof(__bf16));     // 32 MB
  float* partials = (float*)(ws + (size_t)MDIM * KDIM * 2 + (size_t)NDIM * KDIM * 2);
  float* scale = partials + 4096;

  const int nW = NDIM * KDIM;   // 16,777,216
  const int nX = MDIM * KDIM;   // 16,777,216
  const int nW4 = nW / 4;
  const int nX4 = nX / 4;

  absmean_partial_kernel<<<2048, 256, 0, stream>>>(w, partials, nW4);
  absmean_final_kernel<<<1, 256, 0, stream>>>(partials, 2048, scale, 1.f / (float)nW);
  quantize_w_kernel<<<nW4 / 256, 256, 0, stream>>>(w, scale, Wq, nW4);
  convert_x_kernel<<<nX4 / 256, 256, 0, stream>>>(x, Xb, nX4);

  dim3 grid(NDIM / BN, MDIM / BM);   // 32 x 64 blocks
  bitlinear_gemm_kernel<<<grid, 256, 0, stream>>>(Xb, Wq, scale, out);
}
